// SHATTN_55130200212101
// MI455X (gfx1250) — compile-verified, matrix operand order checked by reading, not run on hardware
//
#include <hip/hip_runtime.h>
#include <hip/hip_bf16.h>

// ---------------------------------------------------------------------------
// Single-head causal attention, B=8, T=2048, D=1024, H=64.
// Pass 0: W -> bf16, transposed Wt[3][64][1024] (tiny).
// Pass 1: QKV projection (bf16 WMMA, fp32 accum), double-buffered async
//         LDS staging of Wt chunks -> ws (Q,K row-major bf16, V^T bf16).
// Pass 2: flash-attention (bf16 WMMA, fp32 online softmax) -> d_out fp32.
// ---------------------------------------------------------------------------

typedef __attribute__((ext_vector_type(8)))  float  v8f;
typedef __attribute__((ext_vector_type(16))) __bf16 v16bf;
typedef __attribute__((ext_vector_type(2)))  __bf16 v2bf;
typedef __attribute__((ext_vector_type(4)))  int    v4i;

#define NB   8
#define NT   2048
#define ND   1024
#define NH   64
#define BT   (NB * NT)        // 16384 tokens

#define AS1 __attribute__((address_space(1)))
#define AS3 __attribute__((address_space(3)))

#if __has_builtin(__builtin_amdgcn_global_load_async_to_lds_b128)
#define HAVE_ASYNC_LDS 1
#else
#define HAVE_ASYNC_LDS 0
#endif

#if __has_builtin(__builtin_amdgcn_cvt_pk_bf16_f32)
#define HAVE_CVT_PK 1
#else
#define HAVE_CVT_PK 0
#endif

// ---- fp32 -> bf16 helpers (round-half-up; cheap) --------------------------
__device__ __forceinline__ unsigned fbits(float f) {
    return __builtin_bit_cast(unsigned, f);
}
// pack two fp32 -> two bf16 in one dword (low = a, high = b)
__device__ __forceinline__ unsigned cvtpk(float a, float b) {
#if HAVE_CVT_PK
    v2bf r = __builtin_amdgcn_cvt_pk_bf16_f32(a, b);
    return __builtin_bit_cast(unsigned, r);
#else
    // v_perm_b32: bytes [3:2] = (b+rnd).hi16, [1:0] = (a+rnd).hi16
    return __builtin_amdgcn_perm(fbits(b) + 0x8000u, fbits(a) + 0x8000u,
                                 0x07060302u);
#endif
}
__device__ __forceinline__ __bf16 f2bf(float f) {
    return __builtin_bit_cast(__bf16, (unsigned short)((fbits(f) + 0x8000u) >> 16));
}

__device__ __forceinline__ void wait_async0() {
#if HAVE_ASYNC_LDS
#if __has_builtin(__builtin_amdgcn_s_wait_asynccnt)
    __builtin_amdgcn_s_wait_asynccnt(0);
#else
    asm volatile("s_wait_asynccnt 0" ::: "memory");
#endif
#endif
}

// 16 bf16 lanes-worth of a WMMA A/B operand, viewed as two 16B chunks.
union A16 {
    v16bf v;
    uint4 q[2];
};

__device__ __forceinline__ v8f wmma_bf16(const A16& a, const A16& b, v8f c) {
    return __builtin_amdgcn_wmma_f32_16x16x32_bf16(
        false, a.v, false, b.v, (short)0, c, false, false);
}

// ---------------------------------------------------------------------------
// Kernel 0: convert W[1024,64] fp32 (x3) -> Wt[3][64][1024] bf16 (transposed).
// Reads coalesced over n; scattered bf16 writes (tiny total: 384 KB).
// ---------------------------------------------------------------------------
__global__ __launch_bounds__(256) void SHATTN_convert_w(
    const float* __restrict__ Wq, const float* __restrict__ Wk,
    const float* __restrict__ Wv, __bf16* __restrict__ wt)
{
    int flat = blockIdx.x * 256 + threadIdx.x;      // [0, 3*1024*64)
    int n    = flat & 63;
    int d    = (flat >> 6) & 1023;
    int mat  = flat >> 16;
    const float* W = (mat == 0) ? Wq : ((mat == 1) ? Wk : Wv);
    wt[(size_t)mat * (64 * 1024) + (size_t)n * 1024 + d] = f2bf(W[d * 64 + n]);
}

// ---------------------------------------------------------------------------
// Kernel 1: QKV projection.
// Block = 256 threads (8 waves). Wave w handles tokens [blk*128 + w*16, +16).
// LDS holds double-buffered Wt chunks: [buf][mat*64+n][kk 0..32] bf16 (24 KB),
// staged with async-to-LDS (fallback: uint4 copies) and overlapped with WMMA.
// C layout (per ISA): lane l holds N = l&15, rows M = (l>>4)*8 + j.
// ---------------------------------------------------------------------------
__global__ __launch_bounds__(256) void SHATTN_qkv_proj(
    const float* __restrict__ x, const __bf16* __restrict__ wt,
    const float* __restrict__ bq, const float* __restrict__ bk,
    const float* __restrict__ bv,
    __bf16* __restrict__ qbf, __bf16* __restrict__ kbf, __bf16* __restrict__ vt)
{
    __shared__ __align__(16) __bf16 lds_w[2 * 192 * 32];   // 24 KB

    const int tid   = threadIdx.x;
    const int wave  = tid >> 5;
    const int lane  = tid & 31;
    const int row16 = lane & 15;     // M (A) / N (B) within a 16-wide tile
    const int khalf = lane >> 4;     // selects K-half per ISA 16-bit layout
    const int tokBase = blockIdx.x * 128 + wave * 16;

    // Stage one 12 KB chunk Wt[*][*][d0..d0+32] into lds buffer `buf`.
    // 768 16B segments, 3 per thread. rowi = mat*64+n; Wt row stride = 1024.
    auto stage = [&](int d0, int buf) {
#pragma unroll
        for (int i = 0; i < 3; ++i) {
            int flat = i * 256 + tid;
            int rowi = flat >> 2;
            int seg  = flat & 3;
            const __bf16* gp = wt + (size_t)rowi * 1024 + d0 + seg * 8;
            __bf16* lp = &lds_w[buf * 6144 + rowi * 32 + seg * 8];
#if HAVE_ASYNC_LDS
            __builtin_amdgcn_global_load_async_to_lds_b128(
                (AS1 v4i*)gp, (AS3 v4i*)lp, 0, 0);
#else
            *(uint4*)lp = *(const uint4*)gp;
#endif
        }
    };

    v8f acc[3][4];
#pragma unroll
    for (int m = 0; m < 3; ++m)
#pragma unroll
        for (int n = 0; n < 4; ++n) acc[m][n] = (v8f)0.0f;

    stage(0, 0);
    wait_async0();
    __syncthreads();

    for (int it = 0; it < 32; ++it) {
        const int d0  = it * 32;
        const int buf = it & 1;
        if (it + 1 < 32) stage(d0 + 32, (it + 1) & 1);   // overlap next chunk

        // A operand: x[tokBase+row16, d0 + K(e)]; K(e) = (e&8)*2 + khalf*8 + (e&7)
        A16 a;
        {
            const float* xp = x + (size_t)(tokBase + row16) * ND + d0 + khalf * 8;
            float4 f0 = *(const float4*)(xp);
            float4 f1 = *(const float4*)(xp + 4);
            float4 f2 = *(const float4*)(xp + 16);
            float4 f3 = *(const float4*)(xp + 20);
            a.q[0] = make_uint4(cvtpk(f0.x, f0.y), cvtpk(f0.z, f0.w),
                                cvtpk(f1.x, f1.y), cvtpk(f1.z, f1.w));
            a.q[1] = make_uint4(cvtpk(f2.x, f2.y), cvtpk(f2.z, f2.w),
                                cvtpk(f3.x, f3.y), cvtpk(f3.z, f3.w));
        }

#pragma unroll
        for (int mat = 0; mat < 3; ++mat) {
#pragma unroll
            for (int nt = 0; nt < 4; ++nt) {
                A16 b;
                const __bf16* lp =
                    &lds_w[buf * 6144 + (mat * 64 + nt * 16 + row16) * 32 + khalf * 8];
                b.q[0] = *(const uint4*)(lp);
                b.q[1] = *(const uint4*)(lp + 16);
                acc[mat][nt] = wmma_bf16(a, b, acc[mat][nt]);
            }
        }

        wait_async0();       // next chunk landed
        __syncthreads();     // everyone done reading current chunk
    }

    // Epilogue: bias add, store Q,K row-major bf16 and V transposed bf16.
    const float* bias[3] = { bq, bk, bv };
#pragma unroll
    for (int mat = 0; mat < 3; ++mat) {
#pragma unroll
        for (int nt = 0; nt < 4; ++nt) {
#pragma unroll
            for (int j = 0; j < 8; ++j) {
                int tok = tokBase + khalf * 8 + j;
                int n   = nt * 16 + row16;
                float r = acc[mat][nt][j] + bias[mat][n];
                if (mat == 0)      qbf[(size_t)tok * NH + n] = f2bf(r);
                else if (mat == 1) kbf[(size_t)tok * NH + n] = f2bf(r);
                else               vt[(size_t)n * BT + tok]  = f2bf(r);
            }
        }
    }
}

// ---------------------------------------------------------------------------
// Kernel 2: flash attention over bf16 Q/K/Vt, fp32 online softmax.
// Block = 128 threads (4 waves); block handles 64 queries of one batch,
// wave w handles queries [q0, q0+16), q0 = blockIdx.x*64 + w*16.
// ---------------------------------------------------------------------------
__global__ __launch_bounds__(128) void SHATTN_flash_attn(
    const __bf16* __restrict__ qbf, const __bf16* __restrict__ kbf,
    const __bf16* __restrict__ vt, float* __restrict__ out)
{
    __shared__ __align__(16) __bf16 lds_p[4 * 16 * 32];   // per-wave P buffer

    const int lane  = threadIdx.x & 31;
    const int wave  = threadIdx.x >> 5;
    const int row16 = lane & 15;
    const int khalf = lane >> 4;
    const int q0    = blockIdx.x * 64 + wave * 16;        // query base in batch
    const size_t tokB = (size_t)blockIdx.y * NT;          // batch token base

    // Q in A layout: two 16x32 chunks covering H=64.
    A16 qa[2];
    {
        const __bf16* qp = qbf + (tokB + q0 + row16) * NH;
#pragma unroll
        for (int c = 0; c < 2; ++c) {
            qa[c].q[0] = *(const uint4*)(qp + c * 32 + khalf * 8);
            qa[c].q[1] = *(const uint4*)(qp + c * 32 + 16 + khalf * 8);
        }
    }

    v8f acc0 = (v8f)0.0f, acc1 = (v8f)0.0f, acc2 = (v8f)0.0f, acc3 = (v8f)0.0f;
    float m[8], lsum[8];
#pragma unroll
    for (int j = 0; j < 8; ++j) { m[j] = -1e30f; lsum[j] = 0.0f; }

    __bf16* pbuf = lds_p + wave * 512;
    const int kbEnd = q0 + 16;                            // exclusive key bound

    for (int kb = 0; kb < kbEnd; kb += 32) {
        if (kb + 32 < kbEnd) {
            __builtin_prefetch(kbf + (tokB + kb + 32 + row16) * NH, 0, 1);
        }

        // S tiles: s[t] = Q(16x64) . K[kb+t*16 .. +16]^T  (2 WMMA each)
        v8f s[2];
#pragma unroll
        for (int t = 0; t < 2; ++t) {
            s[t] = (v8f)0.0f;
            const __bf16* kp = kbf + (tokB + kb + t * 16 + row16) * NH;
            A16 b;
            b.q[0] = *(const uint4*)(kp + khalf * 8);
            b.q[1] = *(const uint4*)(kp + 16 + khalf * 8);
            s[t] = wmma_bf16(qa[0], b, s[t]);
            b.q[0] = *(const uint4*)(kp + 32 + khalf * 8);
            b.q[1] = *(const uint4*)(kp + 48 + khalf * 8);
            s[t] = wmma_bf16(qa[1], b, s[t]);
        }

        // Scale, causal mask, online softmax.  Row M = khalf*8 + j lives in
        // lanes sharing `khalf`; reduce with xor shuffles 1,2,4,8.
#pragma unroll
        for (int j = 0; j < 8; ++j) {
            int qg = q0 + khalf * 8 + j;
            int k0 = kb + row16;
            int k1 = kb + 16 + row16;
            float a0 = (k0 <= qg) ? s[0][j] * 0.125f : -1e30f;
            float a1 = (k1 <= qg) ? s[1][j] * 0.125f : -1e30f;

            float mx = fmaxf(a0, a1);
            mx = fmaxf(mx, __shfl_xor(mx, 1));
            mx = fmaxf(mx, __shfl_xor(mx, 2));
            mx = fmaxf(mx, __shfl_xor(mx, 4));
            mx = fmaxf(mx, __shfl_xor(mx, 8));
            float mn = fmaxf(m[j], mx);

            float p0 = __expf(a0 - mn);
            float p1 = __expf(a1 - mn);
            float rs = p0 + p1;
            rs += __shfl_xor(rs, 1);
            rs += __shfl_xor(rs, 2);
            rs += __shfl_xor(rs, 4);
            rs += __shfl_xor(rs, 8);

            float c = __expf(m[j] - mn);
            lsum[j] = lsum[j] * c + rs;
            m[j] = mn;
            acc0[j] *= c; acc1[j] *= c; acc2[j] *= c; acc3[j] *= c;

            // Stash P (16x32) in LDS for re-layout into an A operand.
            int M = khalf * 8 + j;
            pbuf[M * 32 + row16]      = f2bf(p0);
            pbuf[M * 32 + 16 + row16] = f2bf(p1);
        }

        // Per-wave LDS region; DS ops are in-order within a wave, so a
        // dscnt wait (no barrier) is sufficient before reading back.
        asm volatile("s_wait_dscnt 0" ::: "memory");

        A16 pa;
        {
            const __bf16* pp = pbuf + row16 * 32;
            pa.q[0] = *(const uint4*)(pp + khalf * 8);
            pa.q[1] = *(const uint4*)(pp + 16 + khalf * 8);
        }

        // O += P(16x32) . V[kb .. kb+32]; Vt rows give contiguous B chunks.
#pragma unroll
        for (int nt = 0; nt < 4; ++nt) {
            A16 vb;
            const __bf16* vp = vt + (size_t)(nt * 16 + row16) * BT + tokB + kb;
            vb.q[0] = *(const uint4*)(vp + khalf * 8);
            vb.q[1] = *(const uint4*)(vp + 16 + khalf * 8);
            if (nt == 0)      acc0 = wmma_bf16(pa, vb, acc0);
            else if (nt == 1) acc1 = wmma_bf16(pa, vb, acc1);
            else if (nt == 2) acc2 = wmma_bf16(pa, vb, acc2);
            else              acc3 = wmma_bf16(pa, vb, acc3);
        }
    }

    // Normalize and store fp32 output.
#pragma unroll
    for (int j = 0; j < 8; ++j) {
        float inv = 1.0f / lsum[j];
        size_t tok = tokB + q0 + khalf * 8 + j;
        float* op = out + tok * NH;
        op[row16]      = acc0[j] * inv;
        op[16 + row16] = acc1[j] * inv;
        op[32 + row16] = acc2[j] * inv;
        op[48 + row16] = acc3[j] * inv;
    }
}

// ---------------------------------------------------------------------------
extern "C" void kernel_launch(void* const* d_in, const int* in_sizes, int n_in,
                              void* d_out, int out_size, void* d_ws, size_t ws_size,
                              hipStream_t stream) {
    const float* x  = (const float*)d_in[0];
    const float* Wq = (const float*)d_in[1];
    const float* bq = (const float*)d_in[2];
    const float* Wk = (const float*)d_in[3];
    const float* bk = (const float*)d_in[4];
    const float* Wv = (const float*)d_in[5];
    const float* bv = (const float*)d_in[6];

    // Workspace (bf16): Q[BT,64], K[BT,64], Vt[64,BT], Wt[3][64][1024] ~6.7 MB
    __bf16* qbf = (__bf16*)d_ws;
    __bf16* kbf = qbf + (size_t)BT * NH;
    __bf16* vtp = kbf + (size_t)BT * NH;
    __bf16* wtp = vtp + (size_t)BT * NH;

    SHATTN_convert_w<<<dim3(3 * 1024 * 64 / 256), dim3(256), 0, stream>>>(
        Wq, Wk, Wv, wtp);
    SHATTN_qkv_proj<<<dim3(BT / 128), dim3(256), 0, stream>>>(
        x, wtp, bq, bk, bv, qbf, kbf, vtp);
    SHATTN_flash_attn<<<dim3(NT / 64, NB), dim3(128), 0, stream>>>(
        qbf, kbf, vtp, (float*)d_out);
}
